// _BaseMultiHeadAttention_76166950028481
// MI455X (gfx1250) — matrix operand order repaired and checked by reading, not compiled, not run on hardware
//
#include <hip/hip_runtime.h>

// ---------------------------------------------------------------------------
// Causal MHA (B=2, S=2048, H=16, Dh=64) + output projection (1024x1024)
// CDNA5 / gfx1250: wave32, V_WMMA_F32_16X16X32_BF16 everywhere.
// v3: row-sum via WMMA against an all-ones fragment (no LDS shuffle chain),
//     row-max via DPP16 row_ror butterfly (pure VALU, no ds_bpermute).
// ---------------------------------------------------------------------------

typedef __attribute__((ext_vector_type(16))) __bf16 v16bf;
typedef __attribute__((ext_vector_type(8)))  float  v8f;

#define ATTN_B   2
#define ATTN_S   2048
#define ATTN_H   16
#define ATTN_DH  64
#define DMODEL   1024          // H * Dh
#define NEG_INF  (-1.0e9f)
#define QK_SCALE (0.125f)      // 1/sqrt(64)

__device__ __forceinline__ v8f wmma_bf16(v16bf a, v16bf b, v8f c) {
  // D = A(16x32 bf16) * B(32x16 bf16) + C(16x16 f32)
  return __builtin_amdgcn_wmma_f32_16x16x32_bf16(
      /*neg_a=*/false, a, /*neg_b=*/false, b,
      /*c_mod=*/(short)0, c, /*reuse_a=*/false, /*reuse_b=*/false);
}

// Max-reduce across the 16 lanes of a DPP row via row_ror butterfly.
// Stays entirely in VALU: v_mov_b32_dpp + v_max_f32, no LDS round trips.
#define DPP_ROR_MAX(x, n)                                                     \
  (x) = fmaxf((x), __int_as_float(__builtin_amdgcn_update_dpp(                \
            __float_as_int(x), __float_as_int(x), 0x120 + (n), 0xf, 0xf, true)))

__device__ __forceinline__ float rowmax16(float x) {
  DPP_ROR_MAX(x, 8);
  DPP_ROR_MAX(x, 4);
  DPP_ROR_MAX(x, 2);
  DPP_ROR_MAX(x, 1);
  return x;
}

// A-fragment (16x32, row-major source): lane holds row M = lane&15,
// element j holds K = j + 16*(lane>=16).
__device__ __forceinline__ v16bf load_a_frag(const float* __restrict__ row,
                                             int lane, float scale) {
  v16bf a;
  const int kofs = (lane & 16);  // +16 for lanes 16..31
#pragma unroll
  for (int j = 0; j < 16; ++j)
    a[j] = (__bf16)(row[j + kofs] * scale);
  return a;
}

// B-fragment (32x16): lane holds column N = lane&15,
// element j holds K = j + 16*(lane>>4).  Contiguous-in-K source.
__device__ __forceinline__ v16bf load_b_contig(const float* __restrict__ p) {
  v16bf b;
#pragma unroll
  for (int j = 0; j < 16; ++j) b[j] = (__bf16)p[j];
  return b;
}

// Same B-fragment but K-dimension strided in memory (e.g. V: keys x Dh).
__device__ __forceinline__ v16bf load_b_strided(const float* __restrict__ p,
                                                int stride) {
  v16bf b;
#pragma unroll
  for (int j = 0; j < 16; ++j) b[j] = (__bf16)p[(size_t)j * stride];
  return b;
}

// ---------------------------------------------------------------------------
// Kernel 1: flash attention, one 16-row q-tile per wave, 4 waves / block,
// 64 keys per softmax step (4 score tiles -> one m pass).
// Row sums tracked as a 5th WMMA accumulator against an all-ones fragment.
// grid = B*H*32 blocks; wave w handles q-tile (blockIdx&31)*4 + w.
// Writes merged heads (B*S, 1024) f32 into workspace.
// ---------------------------------------------------------------------------
__global__ __launch_bounds__(128) void attn_flash_kernel(
    const float* __restrict__ pre_q, const float* __restrict__ pre_k,
    const float* __restrict__ pre_v, float* __restrict__ merged) {
  const int lane = threadIdx.x & 31;
  const int wave = threadIdx.x >> 5;
  const int lm = lane & 15;   // column (N) / row-within-half
  const int hi = lane >> 4;   // half select

  const int bh = blockIdx.x >> 5;          // 0..31
  const int b  = bh >> 4;
  const int h  = bh & 15;
  const int t  = ((blockIdx.x & 31) << 2) | wave;  // q tile 0..127
  const int qbase = t << 4;

  __shared__ __bf16 pshare[4][16 * 68];    // per-wave P staging (16x64 + pad)
  __bf16* pbuf = &pshare[wave][0];

  const size_t rs = (size_t)DMODEL;        // row stride in (B,S,H,Dh) flat

  // all-ones B fragment: rowsum(P) = P x ones, replicated across N columns
  v16bf ones;
#pragma unroll
  for (int j = 0; j < 16; ++j) ones[j] = (__bf16)1.0f;

  // Q fragments (scale folded in; exact since scale is a power of two)
  const float* qrow = pre_q + ((size_t)(b * ATTN_S) + qbase + lm) * rs + h * ATTN_DH;
  const v16bf qa0 = load_a_frag(qrow, lane, QK_SCALE);
  const v16bf qa1 = load_a_frag(qrow + 32, lane, QK_SCALE);

  v8f acc0 = {}, acc1 = {}, acc2 = {}, acc3 = {}, accl = {};
  float mrow[8];
#pragma unroll
  for (int r = 0; r < 8; ++r) mrow[r] = -3.0e38f;

  // 64 keys per step; rows loaded stay < S because ceil((qbase+16)/64)*64<=2048
  const int nsteps = (qbase + 16 + 63) >> 6;
  for (int step = 0; step < nsteps; ++step) {
    const int k0 = step << 6;

    // ---- scores: four 16x16 tiles (keys k0+16*i .. +15) ----
    v8f sc[4];
#pragma unroll
    for (int i = 0; i < 4; ++i) {
      const float* kp = pre_k + ((size_t)(b * ATTN_S) + k0 + (i << 4) + lm) * rs
                        + h * ATTN_DH + (hi << 4);
      v8f s = {};
      s = wmma_bf16(qa0, load_b_contig(kp), s);
      s = wmma_bf16(qa1, load_b_contig(kp + 32), s);
      sc[i] = s;
    }

    // ---- causal mask (only needed on diagonal steps) ----
    if (k0 + 63 > qbase) {
#pragma unroll
      for (int i = 0; i < 4; ++i) {
        const int key = k0 + (i << 4) + lm;
#pragma unroll
        for (int r = 0; r < 8; ++r) {
          if (key > qbase + r + (hi << 3)) sc[i][r] = NEG_INF;
        }
      }
    }

    // ---- online softmax: one m pass for all 64 keys (DPP row max) ----
#pragma unroll
    for (int r = 0; r < 8; ++r) {
      float tm = fmaxf(fmaxf(sc[0][r], sc[1][r]), fmaxf(sc[2][r], sc[3][r]));
      tm = rowmax16(tm);
      const float mnew  = fmaxf(mrow[r], tm);
      const float alpha = __expf(mrow[r] - mnew);
      mrow[r] = mnew;
      acc0[r] *= alpha; acc1[r] *= alpha; acc2[r] *= alpha; acc3[r] *= alpha;
      accl[r] *= alpha;
      // stage P (C-layout -> LDS, row-major 16x64 bf16, stride 68)
      const int M = r + (hi << 3);
#pragma unroll
      for (int i = 0; i < 4; ++i)
        pbuf[M * 68 + (i << 4) + lm] = (__bf16)__expf(sc[i][r] - mnew);
    }

    // ---- read P back as two A-fragments (same wave: LDS is in-order) ----
    v16bf pa0, pa1;
#pragma unroll
    for (int j = 0; j < 16; ++j) {
      const int kk = j + (hi << 4);
      pa0[j] = pbuf[lm * 68 + kk];
      pa1[j] = pbuf[lm * 68 + 32 + kk];
    }

    // ---- row sums via the matrix core: accl += P x ones ----
    accl = wmma_bf16(pa0, ones, accl);
    accl = wmma_bf16(pa1, ones, accl);

    // ---- P(16x64) x V(64x64): 8 WMMAs over 4 d-subtiles x 2 key-chunks ----
    const float* vp0 = pre_v + ((size_t)(b * ATTN_S) + k0 + (hi << 4)) * rs
                       + h * ATTN_DH + lm;
    const float* vp1 = vp0 + (rs << 5);  // +32 key rows
    acc0 = wmma_bf16(pa0, load_b_strided(vp0 + 0,  DMODEL), acc0);
    acc1 = wmma_bf16(pa0, load_b_strided(vp0 + 16, DMODEL), acc1);
    acc2 = wmma_bf16(pa0, load_b_strided(vp0 + 32, DMODEL), acc2);
    acc3 = wmma_bf16(pa0, load_b_strided(vp0 + 48, DMODEL), acc3);
    acc0 = wmma_bf16(pa1, load_b_strided(vp1 + 0,  DMODEL), acc0);
    acc1 = wmma_bf16(pa1, load_b_strided(vp1 + 16, DMODEL), acc1);
    acc2 = wmma_bf16(pa1, load_b_strided(vp1 + 32, DMODEL), acc2);
    acc3 = wmma_bf16(pa1, load_b_strided(vp1 + 48, DMODEL), acc3);
  }

  // ---- normalize and store merged heads: (b, s, h*64 + d) ----
  float* orow = merged + ((size_t)(b * ATTN_S) + qbase) * DMODEL + h * ATTN_DH;
#pragma unroll
  for (int r = 0; r < 8; ++r) {
    const float inv = 1.0f / accl[r];   // rowsum replicated across lanes
    const int M = r + (hi << 3);
    orow[(size_t)M * DMODEL + lm]      = acc0[r] * inv;
    orow[(size_t)M * DMODEL + lm + 16] = acc1[r] * inv;
    orow[(size_t)M * DMODEL + lm + 32] = acc2[r] * inv;
    orow[(size_t)M * DMODEL + lm + 48] = acc3[r] * inv;
  }
}

// ---------------------------------------------------------------------------
// Kernel 2: out(4096x1024) = X(4096x1024) * W(1024x1024).
// Block = 4 waves sharing an LDS-staged 32x128 bf16 W tile; each wave computes
// a 16x128 output tile (8 WMMAs per K=32 step, A-fragment reused 8x).
// grid = (4096/64 m-groups) * (1024/128 n-groups) = 64*8 = 512 blocks.
// ---------------------------------------------------------------------------
__global__ __launch_bounds__(128) void proj_gemm_kernel(
    const float* __restrict__ X, const float* __restrict__ W,
    float* __restrict__ out) {
  const int lane = threadIdx.x & 31;
  const int wave = threadIdx.x >> 5;
  const int lm = lane & 15;
  const int hi = lane >> 4;

  const int mg = blockIdx.x >> 3;          // 0..63  (64-row group)
  const int ng = blockIdx.x & 7;           // 0..7   (128-col group)
  const int m0 = (mg << 6) + (wave << 4);  // this wave's 16 rows
  const int n0 = ng << 7;

  __shared__ __bf16 wtile[32 * 132];       // 32(K) x 128(N) bf16, stride 132

  v8f acc[8];
#pragma unroll
  for (int i = 0; i < 8; ++i) acc[i] = (v8f){};
  const float* xrow = X + (size_t)(m0 + lm) * DMODEL;

  // cooperative-stage indices: 128 threads cover 32 rows x 4 chunks of 32
  const int skk = threadIdx.x >> 2;        // 0..31 (K row)
  const int sc0 = (threadIdx.x & 3) << 5;  // 0,32,64,96 (N chunk)

  for (int k = 0; k < DMODEL; k += 32) {
    __syncthreads();
    {
      const float* wp = W + (size_t)(k + skk) * DMODEL + n0 + sc0;
#pragma unroll
      for (int j = 0; j < 32; ++j) wtile[skk * 132 + sc0 + j] = (__bf16)wp[j];
    }
    __syncthreads();

    const v16bf a = load_a_frag(xrow + k, lane, 1.0f);
#pragma unroll
    for (int n = 0; n < 8; ++n) {
      v16bf bf;
#pragma unroll
      for (int j = 0; j < 16; ++j)
        bf[j] = wtile[((hi << 4) + j) * 132 + (n << 4) + lm];
      acc[n] = wmma_bf16(a, bf, acc[n]);
    }
  }

  float* orow = out + (size_t)m0 * DMODEL + n0;
#pragma unroll
  for (int n = 0; n < 8; ++n) {
#pragma unroll
    for (int r = 0; r < 8; ++r) {
      const int M = r + (hi << 3);
      orow[(size_t)M * DMODEL + (n << 4) + lm] = acc[n][r];
    }
  }
}

// ---------------------------------------------------------------------------
// setup_inputs order: pre_q, pre_v, pre_k, output_weights, out_seq_len, d_model
// ---------------------------------------------------------------------------
extern "C" void kernel_launch(void* const* d_in, const int* in_sizes, int n_in,
                              void* d_out, int out_size, void* d_ws, size_t ws_size,
                              hipStream_t stream) {
  const float* pre_q = (const float*)d_in[0];
  const float* pre_v = (const float*)d_in[1];
  const float* pre_k = (const float*)d_in[2];
  const float* wproj = (const float*)d_in[3];
  float* out = (float*)d_out;
  float* merged = (float*)d_ws;  // (B*S, 1024) f32 = 16 MB

  // attention: B*H*32 = 1024 blocks, 4 waves each (one q-tile per wave)
  attn_flash_kernel<<<ATTN_B * ATTN_H * 32, 128, 0, stream>>>(pre_q, pre_k, pre_v, merged);
  // projection: 64 m-groups * 8 n-groups = 512 blocks
  proj_gemm_kernel<<<512, 128, 0, stream>>>(merged, wproj, out);
}
